// DeepFM_13030930776067
// MI455X (gfx1250) — matrix operand order repaired and checked by reading, not compiled, not run on hardware
//
#include <hip/hip_runtime.h>
#include <hip/hip_bf16.h>

#define BATCH 16384
#define FDIM  2048
#define KDIM  64

typedef __attribute__((ext_vector_type(16))) __bf16 bf16x16;
typedef __attribute__((ext_vector_type(8)))  float  floatx8;
typedef __attribute__((ext_vector_type(4)))  float  floatx4;
typedef __attribute__((ext_vector_type(8)))  unsigned int uintx8;

__device__ __forceinline__ unsigned short f2bf(float f) {
  unsigned int u = __builtin_bit_cast(unsigned int, f);
  u += 0x7FFFu + ((u >> 16) & 1u);               // round-to-nearest-even
  return (unsigned short)(u >> 16);
}

// Pack two floats into one dword of two bf16 (round-to-nearest, v_perm_b32 pack).
__device__ __forceinline__ unsigned int pack2bf(float a, float b) {
  unsigned int ua = __builtin_bit_cast(unsigned int, a) + 0x8000u;
  unsigned int ub = __builtin_bit_cast(unsigned int, b) + 0x8000u;
  // D[15:0] = ua[31:16], D[31:16] = ub[31:16]
  return __builtin_amdgcn_perm(ub, ua, 0x07060302u);
}

// Prep: vT[64][2048] = bf16 transpose of v[2048][64];  s[i] = sum_k v[i,k]^2
__global__ void fm_prep(const float* __restrict__ v,
                        unsigned short* __restrict__ vT,
                        float* __restrict__ s) {
  int i = blockIdx.x * blockDim.x + threadIdx.x;
  if (i >= FDIM) return;
  const float* vr = v + (size_t)i * KDIM;
  float acc = 0.f;
#pragma unroll
  for (int n = 0; n < KDIM; ++n) {
    float f = vr[n];
    acc += f * f;
    vT[(size_t)n * FDIM + i] = f2bf(f);
  }
  s[i] = acc;
}

// Main: per wave, 16 rows x 64 cols of xv via v_wmma_f32_16x16x32_bf16,
// fused packed-fp32 row-dots for x.w and x^2.s, LDS reduction + sigmoid.
__global__ __launch_bounds__(256) void fm_main(const float* __restrict__ x,
                                               const float* __restrict__ w,
                                               const float* __restrict__ bias,
                                               const unsigned short* __restrict__ vT,
                                               const float* __restrict__ s,
                                               float* __restrict__ out) {
  __shared__ float sm[8][16][17];
  const int lane  = threadIdx.x & 31;
  const int wave  = threadIdx.x >> 5;
  const int mrow  = lane & 15;
  const int half  = lane >> 4;                    // 0 or 1
  const int rowBase = blockIdx.x * 128 + wave * 16;
  const int row   = rowBase + mrow;

  // A-fragment layout: lane<16 -> K 0..15, lane>=16 -> K 16..31
  const float* xr = x + (size_t)row * FDIM + half * 16;
  const float* wr = w + half * 16;
  const float* sr = s + half * 16;
  // B-fragment: lane n holds col (t*16 + n%16), K ko..ko+15 (lanes<16) or +16..31
  const unsigned short* b0r = vT + (size_t)( 0 + mrow) * FDIM + half * 16;
  const unsigned short* b1r = vT + (size_t)(16 + mrow) * FDIM + half * 16;
  const unsigned short* b2r = vT + (size_t)(32 + mrow) * FDIM + half * 16;
  const unsigned short* b3r = vT + (size_t)(48 + mrow) * FDIM + half * 16;

  floatx8 c0 = {}, c1 = {}, c2 = {}, c3 = {};
  floatx4 acc1 = {0.f, 0.f, 0.f, 0.f};            // x . w   partial (packed)
  floatx4 acc2 = {0.f, 0.f, 0.f, 0.f};            // x^2 . s partial (packed)

  for (int ko = 0; ko < FDIM; ko += 32) {
    const floatx4* xp = (const floatx4*)(xr + ko);
    floatx4 f0 = xp[0], f1 = xp[1], f2 = xp[2], f3 = xp[3];
    const floatx4* wp = (const floatx4*)(wr + ko);
    const floatx4* sp = (const floatx4*)(sr + ko);
    floatx4 w0 = wp[0], w1 = wp[1], w2 = wp[2], w3 = wp[3];
    floatx4 s0 = sp[0], s1 = sp[1], s2 = sp[2], s3 = sp[3];

    // packed fma accumulation (v_pk_fma_f32)
    acc1 = f0 * w0 + acc1;
    acc1 = f1 * w1 + acc1;
    acc1 = f2 * w2 + acc1;
    acc1 = f3 * w3 + acc1;
    acc2 = (f0 * f0) * s0 + acc2;
    acc2 = (f1 * f1) * s1 + acc2;
    acc2 = (f2 * f2) * s2 + acc2;
    acc2 = (f3 * f3) * s3 + acc2;

    // A fragment: v_perm_b32 pack, 2 bf16 per dword (K even in [15:0])
    uintx8 au;
    au[0] = pack2bf(f0.x, f0.y);  au[1] = pack2bf(f0.z, f0.w);
    au[2] = pack2bf(f1.x, f1.y);  au[3] = pack2bf(f1.z, f1.w);
    au[4] = pack2bf(f2.x, f2.y);  au[5] = pack2bf(f2.z, f2.w);
    au[6] = pack2bf(f3.x, f3.y);  au[7] = pack2bf(f3.z, f3.w);
    bf16x16 a = __builtin_bit_cast(bf16x16, au);

    bf16x16 b0 = *(const bf16x16*)(b0r + ko);
    bf16x16 b1 = *(const bf16x16*)(b1r + ko);
    bf16x16 b2 = *(const bf16x16*)(b2r + ko);
    bf16x16 b3 = *(const bf16x16*)(b3r + ko);

    c0 = __builtin_amdgcn_wmma_f32_16x16x32_bf16(false, a, false, b0, (short)0, c0, false, false);
    c1 = __builtin_amdgcn_wmma_f32_16x16x32_bf16(false, a, false, b1, (short)0, c1, false, false);
    c2 = __builtin_amdgcn_wmma_f32_16x16x32_bf16(false, a, false, b2, (short)0, c2, false, false);
    c3 = __builtin_amdgcn_wmma_f32_16x16x32_bf16(false, a, false, b3, (short)0, c3, false, false);
  }

  // squared column-sums (v_pk_mul/fma): C layout -> lanes 0..15 hold rows 0..7,
  // lanes 16..31 hold rows 8..15; columns = t*16 + (lane&15)
  floatx8 sqv = c0 * c0;
  sqv = c1 * c1 + sqv;
  sqv = c2 * c2 + sqv;
  sqv = c3 * c3 + sqv;

  // reduce packed dot partials to scalars
  float d1 = acc1.x + acc1.y + acc1.z + acc1.w;
  float d2 = acc2.x + acc2.y + acc2.z + acc2.w;
  // merge K-halves (lane L and L+16 share row L%16)
  d1 += __shfl_xor(d1, 16, 32);
  d2 += __shfl_xor(d2, 16, 32);

#pragma unroll
  for (int j = 0; j < 8; ++j)
    sm[wave][half * 8 + j][mrow] = sqv[j];
  __syncthreads();

  if (lane < 16) {
    float ss = 0.f;
#pragma unroll
    for (int c = 0; c < 16; ++c) ss += sm[wave][lane][c];
    float z = d1 + bias[0] + 0.5f * (ss - d2);
    out[row] = 1.0f / (1.0f + __expf(-z));
  }
}

extern "C" void kernel_launch(void* const* d_in, const int* in_sizes, int n_in,
                              void* d_out, int out_size, void* d_ws, size_t ws_size,
                              hipStream_t stream) {
  const float* x = (const float*)d_in[0];
  const float* w = (const float*)d_in[1];
  const float* b = (const float*)d_in[2];
  const float* v = (const float*)d_in[3];

  unsigned short* vT = (unsigned short*)d_ws;                        // 64*2048*2 = 256 KB
  float* s = (float*)((char*)d_ws + (size_t)KDIM * FDIM * 2);        // + 8 KB

  fm_prep<<<FDIM / 256, 256, 0, stream>>>(v, vT, s);
  fm_main<<<BATCH / 128, 256, 0, stream>>>(x, w, b, vT, s, (float*)d_out);
}
